// TS_CoherAnalysis_32195074851199
// MI455X (gfx1250) — matrix operand order repaired and checked by reading, not compiled, not run on hardware
//
#include <hip/hip_runtime.h>
#include <math.h>

// Problem constants (match reference setup_inputs)
#define B_      32
#define N_      512
#define L_      2048
#define NPERSEG 256
#define STEP    128
#define NSEG    15
#define NF      129
#define NFPAD   144      // 9 tiles of 16
#define NTILES  9
#define KTILES  8        // 256 / 32
#define NREFK   16
#define EPSC    1e-12f
#define TWOPI_OVER_256 (6.28318530717958647692f / 256.0f)

typedef __attribute__((ext_vector_type(16))) _Float16 v16h;
typedef __attribute__((ext_vector_type(8)))  _Float16 v8h;
typedef __attribute__((ext_vector_type(4)))  _Float16 v4h;
typedef __attribute__((ext_vector_type(8)))  float    v8f;

// Workspace layout (bytes, 256-aligned)
#define OFF_CRE    0u        // 8*9*512 f16 = 73728 B
#define OFF_CIM    73728u    // 73728 B
#define OFF_XRE    147456u   // 32*144*16 f32 = 294912 B
#define OFF_XIM    442368u   // 294912 B
#define OFF_PXX    737280u   // 32*144 f32 = 18432 B
#define OFF_SCORES 755712u   // 32*512 f32 = 65536 B
#define OFF_IDX    821248u   // 32*16 int = 2048 B

// ---------------------------------------------------------------------------
// Kernel 1: build DFT matrix C[n,f] = win[n] e^{-2pi i f n/256} - W[f]/256
// stored as f16 directly in WMMA B-fragment order (contiguous K order, same as A frags):
//   frag(kt,nt) flat half index = lane*16 + h, with
//   N = lane%16, K_in_tile = 16*(lane/16) + h, K = kt*32 + K_in_tile
// ---------------------------------------------------------------------------
__global__ void build_dft_frags(_Float16* __restrict__ cre,
                                _Float16* __restrict__ cim) {
    int e = blockIdx.x * blockDim.x + threadIdx.x;   // 0 .. 36863
    if (e >= KTILES * NTILES * 512) return;
    int kt   = e / (NTILES * 512);
    int rem  = e % (NTILES * 512);
    int nt   = rem / 512;
    int le   = rem % 512;             // lane*16 + h
    int lane = le >> 4;
    int h    = le & 15;
    int K = kt * 32 + ((lane >> 4) << 4) + h;   // time index 0..255
    int f = nt * 16 + (lane & 15);              // frequency bin
    float re = 0.f, im = 0.f;
    if (f < NF) {
        float w   = 0.5f - 0.5f * cosf(TWOPI_OVER_256 * (float)K);  // hann periodic
        int   ph  = (K * f) & 255;                                  // arg reduction
        float ang = TWOPI_OVER_256 * (float)ph;
        re = w * cosf(ang);
        im = -w * sinf(ang);
        // detrend fold-in: subtract W[f]/256 (W[0]=128, W[1]=-64, else 0)
        if (f == 0) re -= 0.5f;
        if (f == 1) re += 0.25f;
    }
    cre[e] = (_Float16)re;
    cim[e] = (_Float16)im;
}

// ---------------------------------------------------------------------------
// Kernel 2: target-side detrended/windowed DFT in full f32.
// grid = B_*NSEG blocks; thread t = frequency bin. X stored [b][f][seg].
// ---------------------------------------------------------------------------
__global__ void target_dft(const float* __restrict__ target,
                           float* __restrict__ xre,
                           float* __restrict__ xim) {
    int b   = blockIdx.x / NSEG;
    int seg = blockIdx.x % NSEG;
    __shared__ float xs[NPERSEG];
    int t = threadIdx.x;
    if (t < NPERSEG) xs[t] = target[b * L_ + seg * STEP + t];
    __syncthreads();
    int f = t;
    if (f >= NFPAD) return;
    int o = (b * NFPAD + f) * 16 + seg;
    if (f >= NF) { xre[o] = 0.f; xim[o] = 0.f; return; }
    float mu = 0.f;
    for (int k = 0; k < NPERSEG; ++k) mu += xs[k];
    mu *= (1.0f / NPERSEG);
    float sre = 0.f, sim = 0.f;
    for (int k = 0; k < NPERSEG; ++k) {
        float w   = 0.5f - 0.5f * cosf(TWOPI_OVER_256 * (float)k);
        float v   = (xs[k] - mu) * w;
        int   ph  = (f * k) & 255;
        float ang = TWOPI_OVER_256 * (float)ph;
        sre += v * cosf(ang);
        sim -= v * sinf(ang);
    }
    xre[o] = sre;
    xim[o] = sim;
}

// ---------------------------------------------------------------------------
// Kernel 3: Pxx[b][f] = mean_seg |X|^2 ; also zero the seg=15 padding row.
// ---------------------------------------------------------------------------
__global__ void compute_pxx(float* __restrict__ xre,
                            float* __restrict__ xim,
                            float* __restrict__ pxx) {
    int b = blockIdx.x;
    int f = threadIdx.x;
    if (f >= NFPAD) return;
    int base = (b * NFPAD + f) * 16;
    xre[base + 15] = 0.f;
    xim[base + 15] = 0.f;
    float s = 0.f;
    if (f < NF) {
        for (int sg = 0; sg < NSEG; ++sg) {
            float r = xre[base + sg], i = xim[base + sg];
            s += r * r + i * i;
        }
    }
    pxx[b * NFPAD + f] = s * (1.0f / NSEG);
}

// ---------------------------------------------------------------------------
// Kernel 4: main WMMA kernel. 8 waves per block, one (b,n) series per wave.
// Per series: A (16 segs x 256 samples, f16 via LDS) times DFT B-fragments,
// 144 v_wmma_f32_16x16x32_f16, fused Pyy/Pxy/Cxy reduction -> scores[b][n].
// ---------------------------------------------------------------------------
__global__ __launch_bounds__(256)
void coher_wmma(const float*    __restrict__ db,
                const _Float16* __restrict__ cre_f,
                const _Float16* __restrict__ cim_f,
                const float*    __restrict__ xre,
                const float*    __restrict__ xim,
                const float*    __restrict__ pxx,
                float*          __restrict__ scores) {
    __shared__ _Float16 sx[8][L_];           // 32 KB: one series per wave, f16
    int tid  = threadIdx.x;
    int wave = tid >> 5;
    int lane = tid & 31;
    int series = blockIdx.x * 8 + wave;      // b*512 + n
    int b = series >> 9;

    // Stage series to LDS as f16.  Vectorized: float4 global loads
    // (512 B/wave/instr, HBM-roofline path) + packed 8-byte ds_store_b64.
    const float4* xp4 = (const float4*)(db + (size_t)series * L_);
    for (int i = lane; i < L_ / 4; i += 32) {
        float4 v = xp4[i];
        v4h h4;
        h4[0] = (_Float16)v.x; h4[1] = (_Float16)v.y;
        h4[2] = (_Float16)v.z; h4[3] = (_Float16)v.w;
        *(v4h*)&sx[wave][i * 4] = h4;
    }
    __syncthreads();

    // Build the 8 A-fragments (M=16 segments x K=32) from LDS.
    // A-layout (16-bit, 16x32), contiguous K to match B frags: lane<16 -> M=lane, K = h;
    // lane>=16 -> M=lane-16, K = 16+h. Row M=15 is zero padding (nseg=15).
    int seg = lane & 15;
    int g   = lane >> 4;
    v16h afrag[KTILES];
#pragma unroll
    for (int kt = 0; kt < KTILES; ++kt) {
        v16h a;
        if (seg == 15) {
#pragma unroll
            for (int h = 0; h < 16; ++h) a[h] = (_Float16)0.f;
        } else {
            const _Float16* base = &sx[wave][seg * STEP + kt * 32 + 16 * g];
            v8h r0 = *(const v8h*)(base);        // K = 16g .. 16g+7
            v8h r1 = *(const v8h*)(base + 8);    // K = 16g+8 .. 16g+15
#pragma unroll
            for (int h = 0; h < 8; ++h) { a[h] = r0[h]; a[h + 8] = r1[h]; }
        }
        afrag[kt] = a;
    }

    float score = 0.f;
    int fcol_lo = lane & 15;
    int s0      = g * 8;      // which 8 segment-rows this lane's acc VGPRs hold

    for (int nt = 0; nt < NTILES; ++nt) {
        v8f accre = {};
        v8f accim = {};
#pragma unroll
        for (int kt = 0; kt < KTILES; ++kt) {
            int fo = ((kt * NTILES + nt) << 9) + lane * 16;
            v16h bre = *(const v16h*)(cre_f + fo);
            v16h bim = *(const v16h*)(cim_f + fo);
            accre = __builtin_amdgcn_wmma_f32_16x16x32_f16(
                false, afrag[kt], false, bre, (short)0, accre, false, false);
            accim = __builtin_amdgcn_wmma_f32_16x16x32_f16(
                false, afrag[kt], false, bim, (short)0, accim, false, false);
        }
        if (nt + 1 < NTILES) {
            __builtin_prefetch(cre_f + ((nt + 1) << 9) + lane * 16, 0, 1);
        }

        // C/D layout: lane<16 holds (M=r, N=lane); lane>=16 holds (M=r+8, N=lane-16).
        int fcol = nt * 16 + fcol_lo;
        const float4* xr4 = (const float4*)(xre + (b * NFPAD + fcol) * 16 + s0);
        const float4* xi4 = (const float4*)(xim + (b * NFPAD + fcol) * 16 + s0);
        float4 xr0 = xr4[0], xr1 = xr4[1];
        float4 xi0 = xi4[0], xi1 = xi4[1];
        float xrv[8] = {xr0.x, xr0.y, xr0.z, xr0.w, xr1.x, xr1.y, xr1.z, xr1.w};
        float xiv[8] = {xi0.x, xi0.y, xi0.z, xi0.w, xi1.x, xi1.y, xi1.z, xi1.w};

        float pyy = 0.f, pr = 0.f, pi = 0.f;
#pragma unroll
        for (int r = 0; r < 8; ++r) {
            float yr = accre[r], yi = accim[r];
            pyy += yr * yr + yi * yi;
            pr  += xrv[r] * yr + xiv[r] * yi;   // Re(conj(X)*Y)
            pi  += xrv[r] * yi - xiv[r] * yr;   // Im(conj(X)*Y)
        }
        // fold seg rows 0-7 with 8-15 (lane L <-> L+16)
        pyy += __shfl_xor(pyy, 16);
        pr  += __shfl_xor(pr, 16);
        pi  += __shfl_xor(pi, 16);
        pyy *= (1.0f / NSEG);
        pr  *= (1.0f / NSEG);
        pi  *= (1.0f / NSEG);

        float px  = pxx[b * NFPAD + fcol];
        float cxy = (pr * pr + pi * pi) / (px * pyy + EPSC);
        if (fcol < NF) score += cxy;
    }

    // lanes L and L+16 hold identical values -> reduce within 16-lane halves
    score += __shfl_xor(score, 1);
    score += __shfl_xor(score, 2);
    score += __shfl_xor(score, 4);
    score += __shfl_xor(score, 8);
    if (lane == 0) scores[series] = score * (1.0f / NF);
}

// ---------------------------------------------------------------------------
// Kernel 5: iterative top-16 per batch (tie -> lower index, like lax.top_k)
// ---------------------------------------------------------------------------
__global__ void topk_kernel(const float* __restrict__ scores,
                            int* __restrict__ idx_out) {
    __shared__ float s[N_];
    __shared__ float rv[256];
    __shared__ int   ri[256];
    int b = blockIdx.x, t = threadIdx.x;
    s[t]       = scores[b * N_ + t];
    s[t + 256] = scores[b * N_ + t + 256];
    __syncthreads();
    for (int k = 0; k < NREFK; ++k) {
        float v0 = s[t], v1 = s[t + 256];
        float bv = v0; int bi = t;
        if (v1 > v0) { bv = v1; bi = t + 256; }
        rv[t] = bv; ri[t] = bi;
        __syncthreads();
        for (int off = 128; off > 0; off >>= 1) {
            if (t < off) {
                float ov = rv[t + off]; int oi = ri[t + off];
                if (ov > rv[t] || (ov == rv[t] && oi < ri[t])) { rv[t] = ov; ri[t] = oi; }
            }
            __syncthreads();
        }
        if (t == 0) {
            idx_out[b * NREFK + k] = ri[0];
            s[ri[0]] = -INFINITY;
        }
        __syncthreads();
    }
}

// ---------------------------------------------------------------------------
// Kernel 6: gather selected rows -> d_out [32][16][2048] f32
// ---------------------------------------------------------------------------
__global__ void gather_kernel(const float* __restrict__ db,
                              const int* __restrict__ idx,
                              float* __restrict__ out) {
    int bk = blockIdx.x;           // b*16 + k
    int b  = bk >> 4;
    int n  = idx[bk];
    const float4* src = (const float4*)(db + (size_t)(b * N_ + n) * L_);
    float4*       dst = (float4*)(out + (size_t)bk * L_);
    for (int i = threadIdx.x; i < L_ / 4; i += blockDim.x) dst[i] = src[i];
}

// ---------------------------------------------------------------------------
extern "C" void kernel_launch(void* const* d_in, const int* in_sizes, int n_in,
                              void* d_out, int out_size, void* d_ws, size_t ws_size,
                              hipStream_t stream) {
    const float* target = (const float*)d_in[0];   // [32,1,2048]
    const float* db     = (const float*)d_in[1];   // [32,512,2048]
    float*       out    = (float*)d_out;           // [32,16,2048]

    char* ws = (char*)d_ws;
    _Float16* cre    = (_Float16*)(ws + OFF_CRE);
    _Float16* cim    = (_Float16*)(ws + OFF_CIM);
    float*    xre    = (float*)(ws + OFF_XRE);
    float*    xim    = (float*)(ws + OFF_XIM);
    float*    pxx    = (float*)(ws + OFF_PXX);
    float*    scores = (float*)(ws + OFF_SCORES);
    int*      idx    = (int*)(ws + OFF_IDX);

    // 1. DFT matrix in WMMA B-fragment order (36864 f16 elements per part)
    build_dft_frags<<<144, 256, 0, stream>>>(cre, cim);
    // 2. Target segment DFTs (f32)
    target_dft<<<B_ * NSEG, 256, 0, stream>>>(target, xre, xim);
    // 3. Pxx + padding-row zeroing
    compute_pxx<<<B_, 256, 0, stream>>>(xre, xim, pxx);
    // 4. Main WMMA coherence kernel: 16384 series / 8 per block
    coher_wmma<<<(B_ * N_) / 8, 256, 0, stream>>>(db, cre, cim, xre, xim, pxx, scores);
    // 5. Top-16 per batch
    topk_kernel<<<B_, 256, 0, stream>>>(scores, idx);
    // 6. Gather rows
    gather_kernel<<<B_ * NREFK, 256, 0, stream>>>(db, idx, out);
}
